// TTTLinearLayer_14456859918525
// MI455X (gfx1250) — matrix operand order repaired and checked by reading, not compiled, not run on hardware
//
#include <hip/hip_runtime.h>
#include <math.h>

// ---------------------------------------------------------------------------
// Problem constants (from reference)
// ---------------------------------------------------------------------------
#define Bsz   4
#define Tsz   4096
#define NEsz  1024
#define Hsz   16
#define HKVsz 8
#define Dsz   64
#define CH    64
#define NCH   (Tsz / CH)        // 64
#define BT    (Bsz * Tsz)       // 16384
#define EPSF  1e-6f

// ---------------------------------------------------------------------------
// WMMA types + wrapper (CDNA5 gfx1250, wave32)
// ---------------------------------------------------------------------------
typedef __bf16 v16bf __attribute__((ext_vector_type(16)));
typedef __bf16 v8bf  __attribute__((ext_vector_type(8)));
typedef float  v8f   __attribute__((ext_vector_type(8)));

__device__ __forceinline__ v8f wmma_bf16(v16bf a, v16bf b, v8f c) {
  // D = A(16x32) * B(32x16) + C(16x16), f32 accumulate
  return __builtin_amdgcn_wmma_f32_16x16x32_bf16(
      /*neg_a=*/false, a, /*neg_b=*/false, b,
      /*c_mod=*/(short)0, c, /*reuse_a=*/false, /*reuse_b=*/false);
}

__device__ __forceinline__ v16bf cat8(v8bf lo, v8bf hi) {
  return __builtin_shufflevector(lo, hi, 0, 1, 2, 3, 4, 5, 6, 7,
                                 8, 9, 10, 11, 12, 13, 14, 15);
}

// ---------------------------------------------------------------------------
// Contiguous LDS fragment loaders (each = two ds_load_b128)
// A 16x32 bf16 (row-major source): lane m=row; halves {K..K+7} and {K+16..K+23}
//   (lanes 16-31 take the +8 K offset)
// B 32x16 bf16 from an N-major (transposed) source s[n*ld + k]:
//   lane n=col; halves {K..K+7} and {K+16..K+23} (lanes 16-31 take +8 K offset)
// ---------------------------------------------------------------------------
__device__ __forceinline__ v16bf frag_A(const __bf16* s, int row0, int k0, int ld) {
  int lane = threadIdx.x & 31;
  int m  = row0 + (lane & 15);
  int kb = k0 + ((lane & 16) ? 8 : 0);
  const __bf16* p = s + m * ld + kb;
  return cat8(*(const v8bf*)p, *(const v8bf*)(p + 16));
}

__device__ __forceinline__ v16bf frag_BT(const __bf16* s, int k0, int col0, int ld) {
  int lane = threadIdx.x & 31;
  int n  = col0 + (lane & 15);
  int kb = k0 + ((lane & 16) ? 8 : 0);
  const __bf16* p = s + n * ld + kb;
  return cat8(*(const v8bf*)p, *(const v8bf*)(p + 16));
}

// ---------------------------------------------------------------------------
// Kernel 1a: fp32 -> bf16 conversion (plain)
// ---------------------------------------------------------------------------
__global__ void cvt_kernel(const float* __restrict__ src, __bf16* __restrict__ dst, int n) {
  int i = blockIdx.x * blockDim.x + threadIdx.x;
  if (i < n) dst[i] = (__bf16)src[i];
}

// Kernel 1b: fp32 (K x N) -> bf16 transposed (N x K)
__global__ void cvt_t_kernel(const float* __restrict__ src, __bf16* __restrict__ dst,
                             int K, int N) {
  int gid = blockIdx.x * blockDim.x + threadIdx.x;
  if (gid >= K * N) return;
  int n = gid / K;
  int k = gid - n * K;
  dst[gid] = (__bf16)src[(size_t)k * N + n];
}

// ---------------------------------------------------------------------------
// Kernel 2: QKV projection (bf16 WMMA GEMM) + fused RoPE + RMS epilogue.
// grid.x = BT/64 row tiles, grid.y = 32 head-slots (0..15 q, 16..23 k, 24..31 v)
// Weights come pre-transposed (N x K), staged N-major in LDS.
// ---------------------------------------------------------------------------
__global__ __launch_bounds__(256) void proj_kernel(
    const __bf16* __restrict__ xb,
    const __bf16* __restrict__ wqt, const __bf16* __restrict__ wkt,
    const __bf16* __restrict__ wvt,
    const float* __restrict__ cosT, const float* __restrict__ sinT,
    __bf16* __restrict__ q_ws, __bf16* __restrict__ k_ws, __bf16* __restrict__ v_ws) {
  __shared__ __align__(16) char smem[64 * 64 * 4];  // 16 KB: As+Bs (bf16) reused as Cs (f32)
  __bf16* As = (__bf16*)smem;                 // x tile, row-major  [m][k]
  __bf16* Bs = (__bf16*)(smem + 64 * 64 * 2); // w tile, N-major    [n][k]
  float*  Cs = (float*)smem;

  const int rowTile = blockIdx.x;
  const int slot    = blockIdx.y;
  const int tid     = threadIdx.x;
  const int wave    = tid >> 5;
  const int lane    = tid & 31;

  const __bf16* Wt;
  int colBase;
  if (slot < 16)      { Wt = wqt; colBase = slot * 64; }
  else if (slot < 24) { Wt = wkt; colBase = (slot - 16) * 64; }
  else                { Wt = wvt; colBase = (slot - 24) * 64; }

  const int t0t = wave * 2;
  const int tm0 = t0t >> 2,       tn0 = t0t & 3;
  const int tm1 = (t0t + 1) >> 2, tn1 = (t0t + 1) & 3;

  const int rowBase = rowTile * 64;
  v8f acc0 = {}; v8f acc1 = {};

  for (int kk = 0; kk < NEsz; kk += 64) {
    {
      int idx = tid * 16;
      int r = idx >> 6, c = idx & 63;
      const __bf16* src = xb + (size_t)(rowBase + r) * NEsz + kk + c;
      *(v8bf*)&As[idx]     = *(const v8bf*)src;
      *(v8bf*)&As[idx + 8] = *(const v8bf*)(src + 8);
      // transposed weight: row r = output column (colBase+r), cols = K
      const __bf16* wsrc = Wt + (size_t)(colBase + r) * NEsz + kk + c;
      *(v8bf*)&Bs[idx]     = *(const v8bf*)wsrc;
      *(v8bf*)&Bs[idx + 8] = *(const v8bf*)(wsrc + 8);
    }
    __syncthreads();
#pragma unroll
    for (int ks = 0; ks < 64; ks += 32) {
      acc0 = wmma_bf16(frag_A(As, tm0 * 16, ks, 64), frag_BT(Bs, ks, tn0 * 16, 64), acc0);
      acc1 = wmma_bf16(frag_A(As, tm1 * 16, ks, 64), frag_BT(Bs, ks, tn1 * 16, 64), acc1);
    }
    __syncthreads();
  }

  // spill accumulators (f32) into LDS for row-wise epilogue
  {
    int n  = lane & 15;
    int mb = (lane & 16) ? 8 : 0;
#pragma unroll
    for (int r = 0; r < 8; ++r) Cs[(tm0 * 16 + mb + r) * 64 + tn0 * 16 + n] = acc0[r];
#pragma unroll
    for (int r = 0; r < 8; ++r) Cs[(tm1 * 16 + mb + r) * 64 + tn1 * 16 + n] = acc1[r];
  }
  __syncthreads();

  if (tid < 64) {
    const int r  = tid;
    const int bt = rowBase + r;
    const int b  = bt >> 12;      // / Tsz
    const int t  = bt & (Tsz - 1);
    if (slot < 24) {
      // RoPE then RMS norm (in-place in Cs: each thread owns its row)
      float ss = 0.f;
#pragma unroll
      for (int d = 0; d < 32; ++d) {
        float a  = Cs[r * 64 + d];
        float b2 = Cs[r * 64 + d + 32];
        float cc = cosT[t * 32 + d];
        float sn = sinT[t * 32 + d];
        float o1 = a * cc - b2 * sn;
        float o2 = b2 * cc + a * sn;
        Cs[r * 64 + d]      = o1;
        Cs[r * 64 + d + 32] = o2;
        ss += o1 * o1 + o2 * o2;
      }
      float sc = rsqrtf(ss * (1.f / 64.f) + EPSF);
      if (slot < 16) {
        __bf16* dst = q_ws + (((size_t)b * Hsz + slot) * Tsz + t) * 64;
#pragma unroll
        for (int d = 0; d < 64; ++d) dst[d] = (__bf16)(Cs[r * 64 + d] * sc);
      } else {
        __bf16* dst = k_ws + (((size_t)b * HKVsz + (slot - 16)) * Tsz + t) * 64;
#pragma unroll
        for (int d = 0; d < 64; ++d) dst[d] = (__bf16)(Cs[r * 64 + d] * sc);
      }
    } else {
      __bf16* dst = v_ws + (((size_t)b * HKVsz + (slot - 24)) * Tsz + t) * 64;
#pragma unroll
      for (int d = 0; d < 64; ++d) dst[d] = (__bf16)Cs[r * 64 + d];
    }
  }
}

// ---------------------------------------------------------------------------
// Kernel 3: eta = softplus(x @ lr_w + lr_b), layout [B][HKV][T]
// ---------------------------------------------------------------------------
__global__ void eta_kernel(const float* __restrict__ x, const float* __restrict__ lr_w,
                           const float* __restrict__ lr_b, float* __restrict__ eta_ws) {
  int gid = blockIdx.x * blockDim.x + threadIdx.x;
  if (gid >= BT * HKVsz) return;
  int hkv = gid & (HKVsz - 1);
  int bt  = gid >> 3;
  int b   = bt >> 12;
  int t   = bt & (Tsz - 1);
  const float* xr = x + (size_t)bt * NEsz;
  float acc = lr_b[hkv];
  for (int i = 0; i < NEsz; ++i) acc += xr[i] * lr_w[i * HKVsz + hkv];
  float sp = (acc > 20.f) ? acc : log1pf(expf(acc));
  eta_ws[((size_t)b * HKVsz + hkv) * Tsz + t] = sp;
}

// ---------------------------------------------------------------------------
// Kernel 4: chunked TTT scan. One block per (b,h); W state in LDS (transposed).
//   Wft/Wbt hold W^T  ([e][d])  -> B-frags of k@W, q@W are contiguous
//   VE holds V [c][d], then Es^T [d][c] / [e][c]   (contiguous writes + B-frags)
//   X holds tril(q k^T) [c][e], later K^T [d][c]
// ---------------------------------------------------------------------------
__global__ __launch_bounds__(256) void scan_kernel(
    const __bf16* __restrict__ q_ws, const __bf16* __restrict__ k_ws,
    const __bf16* __restrict__ v_ws, const float* __restrict__ eta_ws,
    const float* __restrict__ onw, const float* __restrict__ w_init_scale,
    __bf16* __restrict__ y_ws) {
  __shared__ float   Wft[64 * 64];  // 16 KB master state, W^T layout
  __shared__ __bf16  Wbt[64 * 64];  //  8 KB bf16 copy, W^T layout
  __shared__ __bf16  Qb[64 * 64];   //  8 KB  [c][d]
  __shared__ __bf16  Kb[64 * 64];   //  8 KB  [c][d]
  __shared__ __bf16  VE[64 * 64];   //  8 KB  V [c][d] -> Es^T [e][c]
  __shared__ __bf16  X[64 * 64];    //  8 KB  Ab [c][e] -> Kt [d][c]
  __shared__ float   etaS[64];
  __shared__ float   rowsq[64];
  __shared__ float   colsq[64];

  const int bh   = blockIdx.x;
  const int b    = bh >> 4;   // / Hsz
  const int h    = bh & 15;
  const int hkv  = h >> 1;    // repeat(.., 2, axis=2)
  const int tid  = threadIdx.x;
  const int wave = tid >> 5;
  const int lane = tid & 31;

  const __bf16* qp = q_ws + ((size_t)b * Hsz + h) * Tsz * 64;
  const __bf16* kp = k_ws + ((size_t)b * HKVsz + hkv) * Tsz * 64;
  const __bf16* vp = v_ws + ((size_t)b * HKVsz + hkv) * Tsz * 64;
  const float*  ep = eta_ws + ((size_t)b * HKVsz + hkv) * Tsz;

  // W0 = I * w_init_scale  (identity: transpose-invariant)
  const float wis = *w_init_scale;
  for (int i = tid; i < 64 * 64; i += 256) {
    float v = ((i >> 6) == (i & 63)) ? wis : 0.f;
    Wft[i] = v;
    Wbt[i] = (__bf16)v;
  }
  __syncthreads();

  const int t0t = wave * 2;
  const int tm0 = t0t >> 2,       tn0 = t0t & 3;
  const int tm1 = (t0t + 1) >> 2, tn1 = (t0t + 1) & 3;
  const int n0  = tn0 * 16 + (lane & 15);
  const int n1  = tn1 * 16 + (lane & 15);
  const int mb0 = tm0 * 16 + ((lane & 16) ? 8 : 0);
  const int mb1 = tm1 * 16 + ((lane & 16) ? 8 : 0);
  const float wcol0 = onw[n0], wcol1 = onw[n1];

  for (int ch = 0; ch < NCH; ++ch) {
    const int t0 = ch * 64;
    {
      int idx = tid * 16;
      size_t g = (size_t)t0 * 64 + idx;
      *(v8bf*)&Qb[idx]     = *(const v8bf*)&qp[g];
      *(v8bf*)&Qb[idx + 8] = *(const v8bf*)&qp[g + 8];
      *(v8bf*)&Kb[idx]     = *(const v8bf*)&kp[g];
      *(v8bf*)&Kb[idx + 8] = *(const v8bf*)&kp[g + 8];
      *(v8bf*)&VE[idx]     = *(const v8bf*)&vp[g];
      *(v8bf*)&VE[idx + 8] = *(const v8bf*)&vp[g + 8];
    }
    if (tid < 64) { etaS[tid] = ep[t0 + tid]; rowsq[tid] = 0.f; colsq[tid] = 0.f; }
    __syncthreads();

    // --- Phase 1 (regs): Es_neg = -eta * (K @ W - V)   [C x D] ---
    float es0[8], es1[8];
    {
      v8f e0 = {}; v8f e1 = {};
#pragma unroll
      for (int ks = 0; ks < 64; ks += 32) {
        e0 = wmma_bf16(frag_A(Kb, tm0 * 16, ks, 64), frag_BT(Wbt, ks, tn0 * 16, 64), e0);
        e1 = wmma_bf16(frag_A(Kb, tm1 * 16, ks, 64), frag_BT(Wbt, ks, tn1 * 16, 64), e1);
      }
#pragma unroll
      for (int r = 0; r < 8; ++r) {
        int m = mb0 + r;
        es0[r] = -etaS[m] * (e0[r] - (float)VE[m * 64 + n0]);
        m = mb1 + r;
        es1[r] = -etaS[m] * (e1[r] - (float)VE[m * 64 + n1]);
      }
    }
    // --- Phase 2 (regs): A = tril(Q @ K^T)   [C x C] ---
    v8f a0 = {}; v8f a1 = {};
#pragma unroll
    for (int ks = 0; ks < 64; ks += 32) {
      a0 = wmma_bf16(frag_A(Qb, tm0 * 16, ks, 64), frag_BT(Kb, ks, tn0 * 16, 64), a0);
      a1 = wmma_bf16(frag_A(Qb, tm1 * 16, ks, 64), frag_BT(Kb, ks, tn1 * 16, 64), a1);
    }
    __syncthreads();  // all V reads complete -> VE reusable

    // Write Es^T into VE ([d][c], contiguous) and A into X ([c][e])
    {
      __bf16* p0 = &VE[n0 * 64 + mb0];
      __bf16* p1 = &VE[n1 * 64 + mb1];
#pragma unroll
      for (int r = 0; r < 8; ++r) { p0[r] = (__bf16)es0[r]; p1[r] = (__bf16)es1[r]; }
#pragma unroll
      for (int r = 0; r < 8; ++r) {
        int m = mb0 + r;
        X[m * 64 + n0] = (__bf16)((n0 <= m) ? a0[r] : 0.f);
        m = mb1 + r;
        X[m * 64 + n1] = (__bf16)((n1 <= m) ? a1[r] : 0.f);
      }
    }
    __syncthreads();

    // --- Phase 3: out = Q@W + A@Es_neg ; RMS rows ; write y (bf16) ---
    v8f o0 = {}; v8f o1 = {};
#pragma unroll
    for (int ks = 0; ks < 64; ks += 32) {
      o0 = wmma_bf16(frag_A(Qb, tm0 * 16, ks, 64), frag_BT(Wbt, ks, tn0 * 16, 64), o0);
      o1 = wmma_bf16(frag_A(Qb, tm1 * 16, ks, 64), frag_BT(Wbt, ks, tn1 * 16, 64), o1);
    }
#pragma unroll
    for (int ks = 0; ks < 64; ks += 32) {
      o0 = wmma_bf16(frag_A(X, tm0 * 16, ks, 64), frag_BT(VE, ks, tn0 * 16, 64), o0);
      o1 = wmma_bf16(frag_A(X, tm1 * 16, ks, 64), frag_BT(VE, ks, tn1 * 16, 64), o1);
    }
#pragma unroll
    for (int r = 0; r < 8; ++r) {
      atomicAdd(&rowsq[mb0 + r], o0[r] * o0[r]);
      atomicAdd(&rowsq[mb1 + r], o1[r] * o1[r]);
    }
    __syncthreads();

    // y writes (uses rowsq) + rebuild X as K^T ([d][c]); Ab is dead now
    {
#pragma unroll
      for (int r = 0; r < 8; ++r) {
        int m = mb0 + r;
        float sc = rsqrtf(rowsq[m] * (1.f / 64.f) + EPSF);
        size_t bt = (size_t)b * Tsz + t0 + m;
        y_ws[bt * (Hsz * Dsz) + h * 64 + n0] = (__bf16)(o0[r] * sc * wcol0);
        m = mb1 + r;
        sc = rsqrtf(rowsq[m] * (1.f / 64.f) + EPSF);
        bt = (size_t)b * Tsz + t0 + m;
        y_ws[bt * (Hsz * Dsz) + h * 64 + n1] = (__bf16)(o1[r] * sc * wcol1);
      }
      // transpose K: X[d][c] = Kb[c][d]; each thread moves 16 elements
      int idx = tid * 16;
      int r2 = idx >> 6, c2 = idx & 63;
      v8bf klo = *(const v8bf*)&Kb[r2 * 64 + c2];
      v8bf khi = *(const v8bf*)&Kb[r2 * 64 + c2 + 8];
#pragma unroll
      for (int j = 0; j < 8; ++j) {
        X[(c2 + j) * 64 + r2]     = klo[j];
        X[(c2 + j + 8) * 64 + r2] = khi[j];
      }
    }
    __syncthreads();

    // --- Phase 4: dW = K^T @ Es_neg  [D x E]; Wn = W + dW/CHUNK; col-normalize.
    // Output element (m=d, n=e) stored into transposed state Wft[e][d]:
    // contiguous per lane. Column norm (over d) == atomicAdd on colsq[n=e].
    v8f u0 = {}; v8f u1 = {};
#pragma unroll
    for (int ks = 0; ks < 64; ks += 32) {
      u0 = wmma_bf16(frag_A(X, tm0 * 16, ks, 64), frag_BT(VE, ks, tn0 * 16, 64), u0);
      u1 = wmma_bf16(frag_A(X, tm1 * 16, ks, 64), frag_BT(VE, ks, tn1 * 16, 64), u1);
    }
    float wn0[8], wn1[8];
    {
      const float* q0 = &Wft[n0 * 64 + mb0];
      const float* q1 = &Wft[n1 * 64 + mb1];
      float s0 = 0.f, s1 = 0.f;
#pragma unroll
      for (int r = 0; r < 8; ++r) {
        wn0[r] = q0[r] + u0[r] * (1.f / (float)CH);
        s0 += wn0[r] * wn0[r];
        wn1[r] = q1[r] + u1[r] * (1.f / (float)CH);
        s1 += wn1[r] * wn1[r];
      }
      atomicAdd(&colsq[n0], s0);
      atomicAdd(&colsq[n1], s1);
    }
    __syncthreads();
    {
      float inv0 = 1.f / fmaxf(sqrtf(colsq[n0]), 1e-12f);
      float inv1 = 1.f / fmaxf(sqrtf(colsq[n1]), 1e-12f);
      float* q0 = &Wft[n0 * 64 + mb0];
      float* q1 = &Wft[n1 * 64 + mb1];
      __bf16* b0 = &Wbt[n0 * 64 + mb0];
      __bf16* b1 = &Wbt[n1 * 64 + mb1];
#pragma unroll
      for (int r = 0; r < 8; ++r) {
        float v0 = wn0[r] * inv0;
        float v1 = wn1[r] * inv1;
        q0[r] = v0; b0[r] = (__bf16)v0;
        q1[r] = v1; b1[r] = (__bf16)v1;
      }
    }
    __syncthreads();
  }
}

// ---------------------------------------------------------------------------
// Kernel 5: out = y @ Wo (bf16 WMMA GEMM, fp32 output); Wo pre-transposed.
// ---------------------------------------------------------------------------
__global__ __launch_bounds__(256) void gemm_out_kernel(
    const __bf16* __restrict__ yb, const __bf16* __restrict__ wot,
    float* __restrict__ out) {
  __shared__ __align__(16) __bf16 As[64 * 64];
  __shared__ __align__(16) __bf16 Bs[64 * 64];   // N-major weight tile

  const int rowTile = blockIdx.x;
  const int colBase = blockIdx.y * 64;
  const int tid     = threadIdx.x;
  const int wave    = tid >> 5;
  const int lane    = tid & 31;

  const int t0t = wave * 2;
  const int tm0 = t0t >> 2,       tn0 = t0t & 3;
  const int tm1 = (t0t + 1) >> 2, tn1 = (t0t + 1) & 3;
  const int rowBase = rowTile * 64;

  v8f acc0 = {}; v8f acc1 = {};
  for (int kk = 0; kk < NEsz; kk += 64) {
    {
      int idx = tid * 16;
      int r = idx >> 6, c = idx & 63;
      const __bf16* src = yb + (size_t)(rowBase + r) * NEsz + kk + c;
      *(v8bf*)&As[idx]     = *(const v8bf*)src;
      *(v8bf*)&As[idx + 8] = *(const v8bf*)(src + 8);
      const __bf16* wsrc = wot + (size_t)(colBase + r) * NEsz + kk + c;
      *(v8bf*)&Bs[idx]     = *(const v8bf*)wsrc;
      *(v8bf*)&Bs[idx + 8] = *(const v8bf*)(wsrc + 8);
    }
    __syncthreads();
#pragma unroll
    for (int ks = 0; ks < 64; ks += 32) {
      acc0 = wmma_bf16(frag_A(As, tm0 * 16, ks, 64), frag_BT(Bs, ks, tn0 * 16, 64), acc0);
      acc1 = wmma_bf16(frag_A(As, tm1 * 16, ks, 64), frag_BT(Bs, ks, tn1 * 16, 64), acc1);
    }
    __syncthreads();
  }
  const int n  = lane & 15;
  const int mb = (lane & 16) ? 8 : 0;
#pragma unroll
  for (int r = 0; r < 8; ++r)
    out[(size_t)(rowBase + tm0 * 16 + mb + r) * NEsz + colBase + tn0 * 16 + n] = acc0[r];
#pragma unroll
  for (int r = 0; r < 8; ++r)
    out[(size_t)(rowBase + tm1 * 16 + mb + r) * NEsz + colBase + tn1 * 16 + n] = acc1[r];
}

// ---------------------------------------------------------------------------
// Host launcher
// ---------------------------------------------------------------------------
extern "C" void kernel_launch(void* const* d_in, const int* in_sizes, int n_in,
                              void* d_out, int out_size, void* d_ws, size_t ws_size,
                              hipStream_t stream) {
  (void)in_sizes; (void)n_in; (void)out_size; (void)ws_size;
  const float* x    = (const float*)d_in[0];
  const float* cosT = (const float*)d_in[1];
  const float* sinT = (const float*)d_in[2];
  const float* Wq   = (const float*)d_in[3];
  const float* Wk   = (const float*)d_in[4];
  const float* Wv   = (const float*)d_in[5];
  const float* Wo   = (const float*)d_in[6];
  const float* lr_w = (const float*)d_in[7];
  const float* lr_b = (const float*)d_in[8];
  const float* onw  = (const float*)d_in[9];
  const float* wis  = (const float*)d_in[10];
  float* out = (float*)d_out;

  char*  ws  = (char*)d_ws;
  size_t off = 0;
  auto alloc = [&](size_t bytes) {
    void* p = ws + off;
    off += (bytes + 255) & ~(size_t)255;
    return p;
  };
  __bf16* x_bf   = (__bf16*)alloc((size_t)BT * NEsz * 2);                    // 32 MB
  __bf16* wqt_bf = (__bf16*)alloc((size_t)NEsz * Hsz * Dsz * 2);             //  2 MB (T)
  __bf16* wkt_bf = (__bf16*)alloc((size_t)NEsz * HKVsz * Dsz * 2);           //  1 MB (T)
  __bf16* wvt_bf = (__bf16*)alloc((size_t)NEsz * HKVsz * Dsz * 2);           //  1 MB (T)
  __bf16* wot_bf = (__bf16*)alloc((size_t)NEsz * NEsz * 2);                  //  2 MB (T)
  __bf16* q_ws   = (__bf16*)alloc((size_t)Bsz * Hsz * Tsz * Dsz * 2);        // 32 MB
  __bf16* k_ws   = (__bf16*)alloc((size_t)Bsz * HKVsz * Tsz * Dsz * 2);      // 16 MB
  __bf16* v_ws   = (__bf16*)alloc((size_t)Bsz * HKVsz * Tsz * Dsz * 2);      // 16 MB
  float*  eta_ws = (float*) alloc((size_t)Bsz * HKVsz * Tsz * 4);            // 0.5 MB
  __bf16* y_ws   = (__bf16*)alloc((size_t)BT * Hsz * Dsz * 2);               // 32 MB

  // 1) bf16 conversions (weights transposed to N x K)
  {
    int n;
    n = BT * NEsz;
    cvt_kernel<<<(n + 255) / 256, 256, 0, stream>>>(x, x_bf, n);
    n = NEsz * Hsz * Dsz;
    cvt_t_kernel<<<(n + 255) / 256, 256, 0, stream>>>(Wq, wqt_bf, NEsz, Hsz * Dsz);
    n = NEsz * HKVsz * Dsz;
    cvt_t_kernel<<<(n + 255) / 256, 256, 0, stream>>>(Wk, wkt_bf, NEsz, HKVsz * Dsz);
    cvt_t_kernel<<<(n + 255) / 256, 256, 0, stream>>>(Wv, wvt_bf, NEsz, HKVsz * Dsz);
    n = NEsz * NEsz;
    cvt_t_kernel<<<(n + 255) / 256, 256, 0, stream>>>(Wo, wot_bf, NEsz, NEsz);
  }
  // 2) QKV projection + RoPE + RMS (WMMA)
  {
    dim3 grid(BT / 64, 32);
    proj_kernel<<<grid, 256, 0, stream>>>(x_bf, wqt_bf, wkt_bf, wvt_bf, cosT, sinT,
                                          q_ws, k_ws, v_ws);
  }
  // 3) eta = softplus(x @ lr_w + lr_b)
  {
    int n = BT * HKVsz;
    eta_kernel<<<(n + 255) / 256, 256, 0, stream>>>(x, lr_w, lr_b, eta_ws);
  }
  // 4) chunked TTT scan (WMMA)
  scan_kernel<<<Bsz * Hsz, 256, 0, stream>>>(q_ws, k_ws, v_ws, eta_ws, onw, wis, y_ws);
  // 5) output projection (WMMA)
  {
    dim3 grid(BT / 64, NEsz / 64);
    gemm_out_kernel<<<grid, 256, 0, stream>>>(y_ws, wot_bf, out);
  }
}
